// BiRNN_20048907337832
// MI455X (gfx1250) — matrix operand order repaired and checked by reading, not compiled, not run on hardware
//
#include <hip/hip_runtime.h>
#include <hip/hip_bf16.h>

// ---------------------------------------------------------------------------
// BiLSTM head for MI455X (gfx1250, wave32, WMMA).
// - out[:, -1] needs the full forward recurrence but only ONE backward step
//   (zero state) -> backward direction = elementwise on xg_b at l=199.
// - Recurrence is independent per batch row -> each workgroup owns 16 rows,
//   h (f16) / c (f32) live in LDS, w_hh stays L2-resident (192 MB L2).
// - All matmuls: v_wmma_f32_16x16x32_f16 (f16 in, f32 accumulate).
// - xg intermediate stored f16 with non-temporal hints: it is the dominant
//   HBM stream (210 MB each way at f16 ~= 18 us/dir @ 23.3 TB/s) and must not
//   evict the hot w_hh / X working set from L2.
// - global_prefetch of the per-step xg slab overlaps the 64-WMMA GEMM.
// ---------------------------------------------------------------------------

#define B_    512
#define L_    200
#define EMB_  128
#define TDIM_ 10
#define IN_   138   // TDIM + EMB
#define KPAD  160   // IN_ padded to 5 chunks of 32
#define H_    256
#define G_    1024  // 4*H

typedef __attribute__((ext_vector_type(16))) _Float16 v16h;
typedef __attribute__((ext_vector_type(8)))  _Float16 v8h;
typedef __attribute__((ext_vector_type(8)))  float    v8f;

union V16U { v16h v; v8h h[2]; };

__device__ __forceinline__ v16h frag2(const _Float16* p0, const _Float16* p1) {
  V16U u;
  u.h[0] = *(const v8h*)p0;
  u.h[1] = *(const v8h*)p1;
  return u.v;
}

__device__ __forceinline__ float sigm_(float x) { return 1.0f / (1.0f + __expf(-x)); }

// --------------------------- kernel 1: ||t||_2 per batch -------------------
__global__ void tnorm_kernel(const int* __restrict__ x, float* __restrict__ nrm) {
  __shared__ float s[128];
  const int b = blockIdx.x, tid = threadIdx.x;
  float acc = 0.f;
  for (int l = tid; l < L_; l += 128) {
    float tv = (float)x[((size_t)b * L_ + l) * 2];
    acc += tv * tv;
  }
  s[tid] = acc;
  __syncthreads();
  for (int off = 64; off; off >>= 1) {
    if (tid < off) s[tid] += s[tid + off];
    __syncthreads();
  }
  if (tid == 0) nrm[b] = sqrtf(s[0]);
}

// ------------------- kernel 2: build f16 input X (B*L, KPAD) ---------------
__global__ void embed_kernel(const int* __restrict__ x,
                             const float* __restrict__ emb,
                             const float* __restrict__ tsW,
                             const float* __restrict__ nrm,
                             _Float16* __restrict__ Xh) {
  const size_t total = (size_t)B_ * L_ * KPAD;
  for (size_t idx = blockIdx.x * (size_t)blockDim.x + threadIdx.x; idx < total;
       idx += (size_t)gridDim.x * blockDim.x) {
    const size_t pos = idx / KPAD;
    const int k = (int)(idx % KPAD);
    float v = 0.f;
    if (k < TDIM_) {
      const int b = (int)(pos / L_);
      const float t = (float)x[pos * 2];
      const float w = tsW[k];
      const float d = fmaxf(fabsf(w) * nrm[b], 1e-12f);
      v = t * w / d;
    } else if (k < IN_) {
      const int id = x[pos * 2 + 1];
      if (id >= 0) v = emb[(size_t)id * EMB_ + (k - TDIM_)];
    }
    Xh[idx] = (_Float16)v;
  }
}

// ----------------- kernel 3: f32 -> f16 weight convert w/ K pad ------------
__global__ void cvt_w(const float* __restrict__ src, _Float16* __restrict__ dst,
                      int rows, int ldS, int ldD) {
  const size_t total = (size_t)rows * ldD;
  for (size_t i = blockIdx.x * (size_t)blockDim.x + threadIdx.x; i < total;
       i += (size_t)gridDim.x * blockDim.x) {
    const int r = (int)(i / ldD), k = (int)(i % ldD);
    dst[i] = (_Float16)(k < ldS ? src[(size_t)r * ldS + k] : 0.f);
  }
}

// -------------- kernel 4: gate GEMM  xg = X @ W_ih^T + b_ih + b_hh ---------
// One 16-row M-tile per block; wave w covers N columns [128w, 128w+128).
// A-fragment layout (ISA 7.12.2, 16-bit A 16x32): lane<16 holds K 0-7,16-23;
// lane>=16 holds K 8-15,24-31 for the same row M=lane&15.
// Output written f16 non-temporal (single-use stream, keep out of L2).
__global__ __launch_bounds__(256) void gate_gemm(
    const _Float16* __restrict__ X, size_t rowStride,   // in halves
    const _Float16* __restrict__ W,                     // [G_][KPAD] f16
    const float* __restrict__ bih, const float* __restrict__ bhh,
    _Float16* __restrict__ out)                         // [M][G_] f16
{
  const int lane = threadIdx.x & 31;
  const int wave = threadIdx.x >> 5;
  const int mt   = blockIdx.x;
  const int m    = lane & 15;
  const int half = lane >> 4;

  const _Float16* xrow = X + (size_t)(mt * 16 + m) * rowStride;
  v16h a[5];
#pragma unroll
  for (int kc = 0; kc < 5; ++kc) {
    const _Float16* p = xrow + kc * 32 + half * 8;
    a[kc] = frag2(p, p + 16);
  }

  const int nbase = wave * 128;
#pragma unroll
  for (int nt = 0; nt < 8; ++nt) {
    const int n0 = nbase + nt * 16;
    v8f acc = {};
#pragma unroll
    for (int kc = 0; kc < 5; ++kc) {
      const _Float16* wp = W + (size_t)(n0 + m) * KPAD + kc * 32 + half * 8;
      v16h bf = frag2(wp, wp + 16);
      acc = __builtin_amdgcn_wmma_f32_16x16x32_f16(false, a[kc], false, bf,
                                                   (short)0, acc, false, false);
    }
    const int col = n0 + m;
    const float bias = bih[col] + bhh[col];
    const size_t row0 = (size_t)mt * 16 + half * 8;
#pragma unroll
    for (int r = 0; r < 8; ++r)
      __builtin_nontemporal_store((_Float16)(acc[r] + bias),
                                  &out[(row0 + r) * G_ + col]);
  }
}

// --------------- kernel 5: forward LSTM recurrence (persistent) ------------
// 32 blocks x 8 waves; block owns 16 batch rows for all 200 steps.
// h (f16) and c (f32) resident in LDS; w_hh tiles streamed from L2.
__global__ __launch_bounds__(256) void lstm_fwd(
    const _Float16* __restrict__ xg,     // [B*L][G_] f16, pos = b*L + l
    const _Float16* __restrict__ Whh,    // [G_][H_] f16
    float* __restrict__ hf_out)          // [B_][H_] f32
{
  __shared__ _Float16 hb[16 * H_];
  __shared__ float    cb[16 * H_];
  const int lane = threadIdx.x & 31;
  const int wave = threadIdx.x >> 5;
  const int m    = lane & 15;
  const int half = lane >> 4;
  const int b0   = blockIdx.x * 16;

  // prefetch geometry: 16 rows x 2 KB of xg per step; 16 threads per row,
  // 256 B (4 cachelines) per thread -> whole slab covered each step.
  const int prow = threadIdx.x >> 4;        // 0..15
  const int pcol = (threadIdx.x & 15) * 128; // halves, 256 B apart

  for (int i = threadIdx.x; i < 16 * H_; i += 256) {
    hb[i] = (_Float16)0.f;
    cb[i] = 0.f;
  }
  __syncthreads();

  for (int t = 0; t < L_; ++t) {
    // kick the step's xg slab toward the WGP while the GEMM runs
    {
      const _Float16* pp = &xg[((size_t)(b0 + prow) * L_ + t) * G_ + pcol];
      __builtin_prefetch(pp, 0, 0);
      __builtin_prefetch(pp + 32, 0, 0);
      __builtin_prefetch(pp + 64, 0, 0);
      __builtin_prefetch(pp + 96, 0, 0);
    }

    // assemble A fragments of h (16 x 256) : 8 K-chunks of 32
    v16h a[8];
#pragma unroll
    for (int kc = 0; kc < 8; ++kc) {
      const _Float16* p = &hb[m * H_ + kc * 32 + half * 8];
      a[kc] = frag2(p, p + 16);
    }
    __syncthreads();  // everyone done reading old h before it is overwritten

    // wave handles hidden columns [32w, 32w+32) across all 4 gates
    v8f acc[8];
#pragma unroll
    for (int nt = 0; nt < 8; ++nt) {
      const int gate = nt >> 1, sub = nt & 1;
      const int col0 = gate * H_ + wave * 32 + sub * 16;
      v8f c = {};
#pragma unroll
      for (int kc = 0; kc < 8; ++kc) {
        const _Float16* wp = Whh + (size_t)(col0 + m) * H_ + kc * 32 + half * 8;
        v16h bf = frag2(wp, wp + 16);
        c = __builtin_amdgcn_wmma_f32_16x16x32_f16(false, a[kc], false, bf,
                                                   (short)0, c, false, false);
      }
      acc[nt] = c;
    }

    // elementwise LSTM cell update
#pragma unroll
    for (int sub = 0; sub < 2; ++sub) {
      const int j = wave * 32 + sub * 16 + m;
#pragma unroll
      for (int r = 0; r < 8; ++r) {
        const int row = r + 8 * half;
        const size_t pos = ((size_t)(b0 + row) * L_ + t) * G_;
        const float xi = acc[0 + sub][r] +
                         (float)__builtin_nontemporal_load(&xg[pos + 0 * H_ + j]);
        const float xf = acc[2 + sub][r] +
                         (float)__builtin_nontemporal_load(&xg[pos + 1 * H_ + j]);
        const float xc = acc[4 + sub][r] +
                         (float)__builtin_nontemporal_load(&xg[pos + 2 * H_ + j]);
        const float xo = acc[6 + sub][r] +
                         (float)__builtin_nontemporal_load(&xg[pos + 3 * H_ + j]);
        const float cprev = cb[row * H_ + j];
        const float cn = sigm_(xf) * cprev + sigm_(xi) * tanhf(xc);
        const float hn = sigm_(xo) * tanhf(cn);
        cb[row * H_ + j] = cn;
        hb[row * H_ + j] = (_Float16)hn;
        if (t == L_ - 1) hf_out[(size_t)(b0 + row) * H_ + j] = hn;
      }
    }
    __syncthreads();  // new h visible before next iteration's reads
  }
}

// -------- kernel 6: backward one-step (zero state) + FC head ---------------
__global__ __launch_bounds__(256) void final_fc(
    const float* __restrict__ hf,        // [B_][H_]
    const _Float16* __restrict__ xgb,    // [B_][G_] f16 (gates at l = L-1)
    const float* __restrict__ fcW,       // [2][2H]
    const float* __restrict__ fcb,       // [2]
    float* __restrict__ out)             // [B_][2]
{
  __shared__ float s0[256], s1[256];
  const int b = blockIdx.x, j = threadIdx.x;
  const float xi = (float)xgb[(size_t)b * G_ + 0 * H_ + j];
  const float xc = (float)xgb[(size_t)b * G_ + 2 * H_ + j];
  const float xo = (float)xgb[(size_t)b * G_ + 3 * H_ + j];
  const float cn  = sigm_(xi) * tanhf(xc);        // f-gate * c_prev(=0) dropped
  const float hbv = sigm_(xo) * tanhf(cn);
  const float hfv = hf[(size_t)b * H_ + j];
  s0[j] = hfv * fcW[j]          + hbv * fcW[H_ + j];
  s1[j] = hfv * fcW[2 * H_ + j] + hbv * fcW[3 * H_ + j];
  __syncthreads();
  for (int off = 128; off; off >>= 1) {
    if (j < off) { s0[j] += s0[j + off]; s1[j] += s1[j + off]; }
    __syncthreads();
  }
  if (j == 0) {
    out[b * 2 + 0] = s0[0] + fcb[0];
    out[b * 2 + 1] = s1[0] + fcb[1];
  }
}

// ---------------------------------------------------------------------------
extern "C" void kernel_launch(void* const* d_in, const int* in_sizes, int n_in,
                              void* d_out, int out_size, void* d_ws, size_t ws_size,
                              hipStream_t stream) {
  const int*   x     = (const int*)d_in[0];
  const float* emb   = (const float*)d_in[1];
  const float* tsW   = (const float*)d_in[2];
  const float* wihf  = (const float*)d_in[3];
  const float* whhf  = (const float*)d_in[4];
  const float* bihf  = (const float*)d_in[5];
  const float* bhhf  = (const float*)d_in[6];
  const float* wihb  = (const float*)d_in[7];
  // d_in[8] = w_hh_b : NOT needed (backward runs exactly one step from h=0)
  const float* bihb  = (const float*)d_in[9];
  const float* bhhb  = (const float*)d_in[10];
  const float* fcW   = (const float*)d_in[11];
  const float* fcb   = (const float*)d_in[12];
  float* out = (float*)d_out;

  // workspace carve-up (256B aligned), ~245 MB total
  char* ws = (char*)d_ws;
  size_t off = 0;
  auto alloc = [&](size_t bytes) {
    void* p = ws + off;
    off += (bytes + 255) & ~(size_t)255;
    return p;
  };
  _Float16* Xh    = (_Float16*)alloc((size_t)B_ * L_ * KPAD * 2);  // 32.8 MB
  _Float16* Wihf  = (_Float16*)alloc((size_t)G_ * KPAD * 2);
  _Float16* Whhf  = (_Float16*)alloc((size_t)G_ * H_ * 2);
  _Float16* Wihb  = (_Float16*)alloc((size_t)G_ * KPAD * 2);
  _Float16* xgf   = (_Float16*)alloc((size_t)B_ * L_ * G_ * 2);    // 210 MB
  _Float16* xgb   = (_Float16*)alloc((size_t)B_ * G_ * 2);
  float*    tnorm = (float*)alloc((size_t)B_ * 4);
  float*    hffin = (float*)alloc((size_t)B_ * H_ * 4);
  (void)ws_size; (void)in_sizes; (void)n_in; (void)out_size;

  // 1) per-batch t-norm
  tnorm_kernel<<<B_, 128, 0, stream>>>(x, tnorm);

  // 2) build f16 input
  embed_kernel<<<8192, 256, 0, stream>>>(x, emb, tsW, tnorm, Xh);

  // 3) weight converts (f32 -> f16, K padded)
  cvt_w<<<640, 256, 0, stream>>>(wihf, Wihf, G_, IN_, KPAD);
  cvt_w<<<1024, 256, 0, stream>>>(whhf, Whhf, G_, H_, H_);
  cvt_w<<<640, 256, 0, stream>>>(wihb, Wihb, G_, IN_, KPAD);

  // 4) gate GEMMs
  //    forward: all (b,l) rows, row-major stride KPAD
  gate_gemm<<<(B_ * L_) / 16, 256, 0, stream>>>(Xh, (size_t)KPAD, Wihf, bihf, bhhf, xgf);
  //    backward: only l = L-1, one row per batch (stride L*KPAD)
  gate_gemm<<<B_ / 16, 256, 0, stream>>>(Xh + (size_t)(L_ - 1) * KPAD,
                                         (size_t)L_ * KPAD, Wihb, bihb, bhhb, xgb);

  // 5) forward recurrence (200 sequential steps, per-batch-row independent)
  lstm_fwd<<<B_ / 16, 256, 0, stream>>>(xgf, Whhf, hffin);

  // 6) backward one-step + FC head
  final_fc<<<B_, 256, 0, stream>>>(hffin, xgb, fcW, fcb, out);
}
